// TransformerDecoder_24300924960866
// MI455X (gfx1250) — matrix operand order repaired and checked by reading, not compiled, not run on hardware
//
#include <hip/hip_runtime.h>

// ---------------------------------------------------------------------------
// Problem constants (match the reference)
// ---------------------------------------------------------------------------
constexpr int Lc  = 6;
constexpr int Bc  = 4;
constexpr int Tc  = 1024;
constexpr int Sc  = 1024;
constexpr int Dc  = 1024;
constexpr int Hc  = 16;
constexpr int HDc = 64;
constexpr float SCALE_C = 0.125f;   // 1/sqrt(64)
constexpr float EPSC    = 1e-5f;

typedef __bf16 bf16_t;
typedef __attribute__((ext_vector_type(16))) __bf16 v16bf;
typedef __attribute__((ext_vector_type(8)))  __bf16 v8bf;
typedef __attribute__((ext_vector_type(8)))  float  v8f;
typedef __attribute__((ext_vector_type(4)))  unsigned int v4u;
typedef __attribute__((ext_vector_type(8)))  int v8i;
typedef __attribute__((ext_vector_type(4)))  int v4i;

__device__ __forceinline__ v8f wmma_bf16(v16bf a, v16bf b, v8f c) {
  // D = A(16x32 bf16) * B(32x16 bf16) + C(16x16 f32)
  return __builtin_amdgcn_wmma_f32_16x16x32_bf16(
      /*neg_a=*/false, a, /*neg_b=*/false, b,
      /*c_mod=*/(short)0, c, /*reuse_a=*/false, /*reuse_b=*/false);
}

// ---------------------------------------------------------------------------
// Tensor Data Mover helpers (CDNA5 D# per CDNA5 ISA)
// 2D tile: data_size = 4B units; tile = 16 dwords (64B) x 128 rows;
// row stride = row_dwords; group2/group3 zero (2D tensor).
// This toolchain exposes the 6-arg builtin:
//   (uint32x4 g0, int32x8 g1, int32x4 g2, int32x4 g3, int32x8, i32 cpol)
// ---------------------------------------------------------------------------
__device__ __forceinline__ v8i make_tdm_g1(int rows, int row_dwords) {
  v8i g1;
  g1[0] = (2 << 16);                                        // data_size=2 (4B)
  g1[1] = (row_dwords & 0xFFFF) << 16;                      // tensor_dim0[15:0]
  g1[2] = ((row_dwords >> 16) & 0xFFFF) | ((rows & 0xFFFF) << 16); // dim0 hi | dim1 lo
  g1[3] = ((rows >> 16) & 0xFFFF) | (16 << 16);             // dim1 hi | tile_dim0=16
  g1[4] = 128;                                              // tile_dim1=128, tile_dim2=0
  g1[5] = row_dwords;                                       // tensor_dim0_stride lo32
  g1[6] = 0;
  g1[7] = 0;
  return g1;
}

__device__ __forceinline__ void tdm_load_tile(const bf16_t* gbase, unsigned lds_off, v8i g1) {
  const unsigned long long ga = (unsigned long long)(uintptr_t)gbase;  // 57-bit
  v4u g0;
  g0[0] = 1u;                                 // count=1, user descriptor
  g0[1] = lds_off;                            // lds_addr (bytes)
  g0[2] = (unsigned)ga;                       // global_addr[31:0]
  g0[3] = (unsigned)(ga >> 32) | (2u << 30);  // global_addr[56:32] | type=2
  const v4i z4 = {0, 0, 0, 0};
  const v8i z8 = {0, 0, 0, 0, 0, 0, 0, 0};
  __builtin_amdgcn_tensor_load_to_lds(g0, g1, z4, z4, z8, 0);
}

// ---------------------------------------------------------------------------
// Conversion / copy kernels
// ---------------------------------------------------------------------------
__global__ __launch_bounds__(256) void cvt_kernel(const float* __restrict__ in,
                                                  bf16_t* __restrict__ out,
                                                  size_t n) {
  size_t i = (size_t)blockIdx.x * 256 + threadIdx.x;
  if (i < n) out[i] = (bf16_t)in[i];
}

__global__ __launch_bounds__(256) void copy_cvt_kernel(const float* __restrict__ in,
                                                       float* __restrict__ fout,
                                                       bf16_t* __restrict__ bout,
                                                       size_t n) {
  size_t i = (size_t)blockIdx.x * 256 + threadIdx.x;
  if (i < n) {
    float v = in[i];
    fout[i] = v;
    bout[i] = (bf16_t)v;
  }
}

// ---------------------------------------------------------------------------
// bf16 WMMA GEMM:  C[M,N] = A[M,K] @ W[N,K]^T + bias[N]
//   EPI bit0: write f32 to Cf, bit1: write bf16 to Cb, bit2: exact GELU
// Block: 256 threads (8 waves); block tile 128(M) x 128(N), BK=32.
// Wave grid 4(M) x 2(N): each wave 32x64 = 2 A-frags x 4 B-frags = 8 WMMAs.
// Tile fills via TDM (tensor_load_to_lds), double buffered, one barrier/step:
//   wave0: s_wait_tensorcnt 0 -> barrier -> issue next tile into buf^1 ->
//   all waves compute buf (reads of buf^1 finished before previous barrier).
// ---------------------------------------------------------------------------
template <int EPI>
__global__ __launch_bounds__(256)
void gemm_bf16(const bf16_t* __restrict__ A, const bf16_t* __restrict__ W,
               const float* __restrict__ bias,
               float* __restrict__ Cf, bf16_t* __restrict__ Cb,
               int M, int N, int K) {
  __shared__ bf16_t At[2][128][32];
  __shared__ bf16_t Wt[2][128][32];

  const int tid  = threadIdx.x;
  const int wave = tid >> 5;
  const int lane = tid & 31;
  const int g    = lane >> 4;     // 16-lane group
  const int ln16 = lane & 15;

  const int m0 = blockIdx.y * 128;
  const int n0 = blockIdx.x * 128;
  const int wm = (wave & 3) * 32;   // wave M offset in tile
  const int wn = (wave >> 2) * 64;  // wave N offset in tile

  v8f acc[2][4];
  {
    v8f z = {};
#pragma unroll
    for (int s = 0; s < 2; s++)
#pragma unroll
      for (int nb = 0; nb < 4; nb++) acc[s][nb] = z;
  }

  const v8i g1a = make_tdm_g1(M, K >> 1);
  const v8i g1w = make_tdm_g1(N, K >> 1);
  const unsigned ldsA[2] = {(unsigned)(uintptr_t)&At[0][0][0],
                            (unsigned)(uintptr_t)&At[1][0][0]};
  const unsigned ldsW[2] = {(unsigned)(uintptr_t)&Wt[0][0][0],
                            (unsigned)(uintptr_t)&Wt[1][0][0]};

  if (wave == 0) {
    tdm_load_tile(A + (size_t)m0 * K, ldsA[0], g1a);
    tdm_load_tile(W + (size_t)n0 * K, ldsW[0], g1w);
  }

  int buf = 0;
  for (int k0 = 0; k0 < K; k0 += 32) {
    if (wave == 0) __builtin_amdgcn_s_wait_tensorcnt(0);
    __syncthreads();
    if (wave == 0 && k0 + 32 < K) {
      tdm_load_tile(A + (size_t)m0 * K + (k0 + 32), ldsA[buf ^ 1], g1a);
      tdm_load_tile(W + (size_t)n0 * K + (k0 + 32), ldsW[buf ^ 1], g1w);
    }

    // A fragments (16x32): lane group g holds 16 contiguous K values at g*16
    v16bf af[2];
#pragma unroll
    for (int s = 0; s < 2; s++) {
      const bf16_t* ar = &At[buf][wm + s * 16 + ln16][0];
      v8bf lo = *(const v8bf*)(ar + g * 16);
      v8bf hi = *(const v8bf*)(ar + g * 16 + 8);
#pragma unroll
      for (int i = 0; i < 8; i++) { af[s][i] = lo[i]; af[s][8 + i] = hi[i]; }
    }
    // B fragments (32x16): lane = N, 16 contiguous K values at g*16;
    // each B-fragment reused across both A-fragments.
#pragma unroll
    for (int nb = 0; nb < 4; nb++) {
      v16bf bfr = *(const v16bf*)&Wt[buf][wn + nb * 16 + ln16][g * 16];
      acc[0][nb] = wmma_bf16(af[0], bfr, acc[0][nb]);
      acc[1][nb] = wmma_bf16(af[1], bfr, acc[1][nb]);
    }
    buf ^= 1;
  }

  // Epilogue: C layout -> row = r + 8*g, col = ln16
#pragma unroll
  for (int s = 0; s < 2; s++) {
    const int row0 = m0 + wm + s * 16 + g * 8;
#pragma unroll
    for (int nb = 0; nb < 4; nb++) {
      const int col = n0 + wn + nb * 16 + ln16;
      const float bv = bias[col];
#pragma unroll
      for (int r = 0; r < 8; r++) {
        float v = acc[s][nb][r] + bv;
        if (EPI & 4) v = 0.5f * v * (1.0f + erff(v * 0.70710678118f));
        const size_t idx = (size_t)(row0 + r) * N + col;
        if (EPI & 1) Cf[idx] = v;
        if (EPI & 2) Cb[idx] = (bf16_t)v;
      }
    }
  }
}

// ---------------------------------------------------------------------------
// Flash attention (bf16 WMMA, fp32 online softmax).
// Q,K,V,O layout: [(b*T + t)*D + h*HD + d]  (head-interleaved, bf16)
// Grid: (Tq/128, B*H); block = 256 threads (8 waves), each wave = 16 queries.
// Key loop in chunks of 32; V staged transposed for contiguous PV B-frags.
// ---------------------------------------------------------------------------
template <bool CAUSAL>
__global__ __launch_bounds__(256)
void flash_attn(const bf16_t* __restrict__ Q, const bf16_t* __restrict__ Kb,
                const bf16_t* __restrict__ Vb, bf16_t* __restrict__ O,
                int Tq, int Sk) {
  __shared__ bf16_t Kt[32][64];        // [key][d]
  __shared__ bf16_t Vt[64][32];        // [d][key]  (transposed)
  __shared__ bf16_t Pl[8][16][32];     // per-wave P tile staging

  const int tid  = threadIdx.x;
  const int wave = tid >> 5;
  const int lane = tid & 31;
  const int g    = lane >> 4;
  const int ln16 = lane & 15;

  const int bh = blockIdx.y;
  const int b  = bh / Hc;
  const int h  = bh % Hc;
  const int q0 = blockIdx.x * 128 + wave * 16;

  // Q fragments for d = [0,32) and [32,64)
  const bf16_t* qrow = Q + ((size_t)(b * Tq + q0 + ln16)) * Dc + h * HDc;
  v16bf qf[2];
#pragma unroll
  for (int c = 0; c < 2; c++) {
    v8bf lo = *(const v8bf*)(qrow + c * 32 + g * 16);
    v8bf hi = *(const v8bf*)(qrow + c * 32 + g * 16 + 8);
#pragma unroll
    for (int i = 0; i < 8; i++) { qf[c][i] = lo[i]; qf[c][8 + i] = hi[i]; }
  }

  v8f o[4];
  {
    v8f z = {};
#pragma unroll
    for (int dt = 0; dt < 4; dt++) o[dt] = z;
  }
  float mrow[8], lrow[8];
#pragma unroll
  for (int r = 0; r < 8; r++) { mrow[r] = -1e30f; lrow[r] = 0.0f; }

  const int sEnd = CAUSAL ? min(Sk, (int)blockIdx.x * 128 + 128) : Sk;

  for (int s0 = 0; s0 < sEnd; s0 += 32) {
    // cooperative K/V staging: thread -> (key = tid>>3, 8 dims)
    {
      const int key  = tid >> 3;
      const int dseg = (tid & 7) * 8;
      const size_t base = ((size_t)(b * Sk + s0 + key)) * Dc + h * HDc + dseg;
      *(v8bf*)&Kt[key][dseg] = *(const v8bf*)&Kb[base];
      v8bf vv = *(const v8bf*)&Vb[base];
#pragma unroll
      for (int i = 0; i < 8; i++) Vt[dseg + i][key] = vv[i];
      if (s0 + 32 < sEnd) {  // near-prefetch of next key chunk
        __builtin_prefetch(&Kb[base + 32 * Dc], 0, 3);
        __builtin_prefetch(&Vb[base + 32 * Dc], 0, 3);
      }
    }
    __syncthreads();

    // scores for two 16-key sub-tiles (contract over d=64 via 2 WMMAs each)
    float sc[2][8];
#pragma unroll
    for (int kt = 0; kt < 2; kt++) {
      v8f st = {};
#pragma unroll
      for (int c = 0; c < 2; c++) {
        v16bf kf = *(const v16bf*)&Kt[kt * 16 + ln16][c * 32 + g * 16];
        st = wmma_bf16(qf[c], kf, st);
      }
#pragma unroll
      for (int r = 0; r < 8; r++) {
        float v = st[r] * SCALE_C;
        if (CAUSAL) {
          const int qg = q0 + r + 8 * g;
          const int kg = s0 + kt * 16 + ln16;
          if (kg > qg) v = -1e30f;
        }
        sc[kt][r] = v;
      }
    }

    // online softmax (row = query, spread over the 16-lane half)
#pragma unroll
    for (int r = 0; r < 8; r++) {
      float rm = fmaxf(sc[0][r], sc[1][r]);
#pragma unroll
      for (int off = 8; off > 0; off >>= 1) rm = fmaxf(rm, __shfl_xor(rm, off, 32));
      const float mn  = fmaxf(mrow[r], rm);
      const float fac = __expf(mrow[r] - mn);
      const float p0  = __expf(sc[0][r] - mn);
      const float p1  = __expf(sc[1][r] - mn);
      float ps = p0 + p1;
#pragma unroll
      for (int off = 8; off > 0; off >>= 1) ps += __shfl_xor(ps, off, 32);
      lrow[r] = lrow[r] * fac + ps;
      mrow[r] = mn;
#pragma unroll
      for (int dt = 0; dt < 4; dt++) o[dt][r] *= fac;
      Pl[wave][r + 8 * g][ln16]      = (bf16_t)p0;
      Pl[wave][r + 8 * g][16 + ln16] = (bf16_t)p1;
    }

    // P as A-fragment (16x32 over 32 keys), then PV: 4 WMMAs for d=64
    v16bf pf;
    {
      const bf16_t* pr = &Pl[wave][ln16][0];
      v8bf lo = *(const v8bf*)(pr + g * 16);
      v8bf hi = *(const v8bf*)(pr + g * 16 + 8);
#pragma unroll
      for (int i = 0; i < 8; i++) { pf[i] = lo[i]; pf[8 + i] = hi[i]; }
    }
#pragma unroll
    for (int dt = 0; dt < 4; dt++) {
      v16bf vf = *(const v16bf*)&Vt[dt * 16 + ln16][g * 16];
      o[dt] = wmma_bf16(pf, vf, o[dt]);
    }
    __syncthreads();
  }

  // normalize and write bf16 output
#pragma unroll
  for (int r = 0; r < 8; r++) {
    const float inv = 1.0f / lrow[r];
    const int q = q0 + r + 8 * g;
    bf16_t* orow = O + ((size_t)(b * Tq + q)) * Dc + h * HDc;
#pragma unroll
    for (int dt = 0; dt < 4; dt++)
      orow[dt * 16 + ln16] = (bf16_t)(o[dt][r] * inv);
  }
}

// ---------------------------------------------------------------------------
// LayerNorm helpers/kernels (fp32). One block (256 threads) per row of D=1024.
// ---------------------------------------------------------------------------
__device__ __forceinline__ float block_reduce_sum(float v, float* sbuf) {
#pragma unroll
  for (int off = 16; off > 0; off >>= 1) v += __shfl_xor(v, off, 32);
  const int wave = threadIdx.x >> 5, lane = threadIdx.x & 31;
  __syncthreads();
  if (lane == 0) sbuf[wave] = v;
  __syncthreads();
  float s = 0.f;
#pragma unroll
  for (int i = 0; i < 8; i++) s += sbuf[i];
  return s;
}

// x = LN(x + y; w, b)   (writes fp32 x and bf16 copy xb)
__global__ __launch_bounds__(256)
void add_ln_kernel(float* __restrict__ x, const float* __restrict__ y,
                   const float* __restrict__ w, const float* __restrict__ b,
                   bf16_t* __restrict__ xb) {
  __shared__ float sbuf[8];
  const size_t row = blockIdx.x;
  float* xr = x + row * Dc;
  const float* yr = y + row * Dc;
  float v[4]; float s = 0.f;
#pragma unroll
  for (int i = 0; i < 4; i++) {
    const int c = threadIdx.x + 256 * i;
    v[i] = xr[c] + yr[c];
    s += v[i];
  }
  const float mean = block_reduce_sum(s, sbuf) * (1.0f / Dc);
  float vs = 0.f;
#pragma unroll
  for (int i = 0; i < 4; i++) { const float d = v[i] - mean; vs += d * d; }
  const float rs = rsqrtf(block_reduce_sum(vs, sbuf) * (1.0f / Dc) + EPSC);
#pragma unroll
  for (int i = 0; i < 4; i++) {
    const int c = threadIdx.x + 256 * i;
    const float o = (v[i] - mean) * rs * w[c] + b[c];
    xr[c] = o;
    xb[row * Dc + c] = (bf16_t)o;
  }
}

// FFN tail per reference: ff = x + LN(y; fnw, fnb); x = LN(x + ff; l3w, l3b)
//                         => x = LN(2*x + LN(y); l3w, l3b)
__global__ __launch_bounds__(256)
void ffn_ln_kernel(float* __restrict__ x, const float* __restrict__ y,
                   const float* __restrict__ fnw, const float* __restrict__ fnb,
                   const float* __restrict__ l3w, const float* __restrict__ l3b,
                   bf16_t* __restrict__ xb) {
  __shared__ float sbuf[8];
  const size_t row = blockIdx.x;
  float* xr = x + row * Dc;
  const float* yr = y + row * Dc;
  float yv[4]; float s = 0.f;
#pragma unroll
  for (int i = 0; i < 4; i++) {
    const int c = threadIdx.x + 256 * i;
    yv[i] = yr[c]; s += yv[i];
  }
  const float m1 = block_reduce_sum(s, sbuf) * (1.0f / Dc);
  float vs = 0.f;
#pragma unroll
  for (int i = 0; i < 4; i++) { const float d = yv[i] - m1; vs += d * d; }
  const float r1 = rsqrtf(block_reduce_sum(vs, sbuf) * (1.0f / Dc) + EPSC);
  float tv[4]; s = 0.f;
#pragma unroll
  for (int i = 0; i < 4; i++) {
    const int c = threadIdx.x + 256 * i;
    tv[i] = 2.0f * xr[c] + (yv[i] - m1) * r1 * fnw[c] + fnb[c];
    s += tv[i];
  }
  const float m2 = block_reduce_sum(s, sbuf) * (1.0f / Dc);
  vs = 0.f;
#pragma unroll
  for (int i = 0; i < 4; i++) { const float d = tv[i] - m2; vs += d * d; }
  const float r2 = rsqrtf(block_reduce_sum(vs, sbuf) * (1.0f / Dc) + EPSC);
#pragma unroll
  for (int i = 0; i < 4; i++) {
    const int c = threadIdx.x + 256 * i;
    const float o = (tv[i] - m2) * r2 * l3w[c] + l3b[c];
    xr[c] = o;
    xb[row * Dc + c] = (bf16_t)o;
  }
}

// ---------------------------------------------------------------------------
// Orchestration
// ---------------------------------------------------------------------------
extern "C" void kernel_launch(void* const* d_in, const int* in_sizes, int n_in,
                              void* d_out, int out_size, void* d_ws, size_t ws_size,
                              hipStream_t stream) {
  (void)in_sizes; (void)n_in; (void)out_size; (void)ws_size;

  const float* x_in = (const float*)d_in[0];
  const float* enc  = (const float*)d_in[1];
  // weight order: sq, sk, sv, so, cq, ck, cv, co, f1, f2
  const float* Wsrc[10] = {
    (const float*)d_in[2],  (const float*)d_in[4],  (const float*)d_in[6],
    (const float*)d_in[8],  (const float*)d_in[10], (const float*)d_in[12],
    (const float*)d_in[14], (const float*)d_in[16], (const float*)d_in[18],
    (const float*)d_in[20]
  };
  const float* Bsrc[10] = {
    (const float*)d_in[3],  (const float*)d_in[5],  (const float*)d_in[7],
    (const float*)d_in[9],  (const float*)d_in[11], (const float*)d_in[13],
    (const float*)d_in[15], (const float*)d_in[17], (const float*)d_in[19],
    (const float*)d_in[21]
  };
  const float* fnw = (const float*)d_in[22];
  const float* fnb = (const float*)d_in[23];
  const float* l1w = (const float*)d_in[24];
  const float* l1b = (const float*)d_in[25];
  const float* l2w = (const float*)d_in[26];
  const float* l2b = (const float*)d_in[27];
  const float* l3w = (const float*)d_in[28];
  const float* l3b = (const float*)d_in[29];

  float* x = (float*)d_out;  // running fp32 activations; final value = output

  // workspace carve-out
  char* p = (char*)d_ws;
  auto carve = [&p](size_t bytes) -> void* {
    void* r = (void*)p;
    p += (bytes + 255) & ~(size_t)255;
    return r;
  };
  const size_t wN   = (size_t)Lc * Dc * Dc;       // elems per weight tensor
  const size_t actN = (size_t)Bc * Tc * Dc;       // elems per activation
  bf16_t* Wb   = (bf16_t*)carve(10 * wN * sizeof(bf16_t));
  bf16_t* xb   = (bf16_t*)carve(actN * sizeof(bf16_t));
  bf16_t* encb = (bf16_t*)carve(actN * sizeof(bf16_t));
  bf16_t* qb   = (bf16_t*)carve(actN * sizeof(bf16_t));
  bf16_t* kb   = (bf16_t*)carve(actN * sizeof(bf16_t));
  bf16_t* vb   = (bf16_t*)carve(actN * sizeof(bf16_t));
  bf16_t* ab   = (bf16_t*)carve(actN * sizeof(bf16_t));
  bf16_t* hb   = (bf16_t*)carve(actN * sizeof(bf16_t));
  float*  t0   = (float*)carve(actN * sizeof(float));

  // one-time (per call) conversions
  for (int wi = 0; wi < 10; wi++)
    cvt_kernel<<<(unsigned)((wN + 255) / 256), 256, 0, stream>>>(
        Wsrc[wi], Wb + wi * wN, wN);
  copy_cvt_kernel<<<(unsigned)((actN + 255) / 256), 256, 0, stream>>>(
      x_in, x, xb, actN);
  cvt_kernel<<<(unsigned)((actN + 255) / 256), 256, 0, stream>>>(enc, encb, actN);

  const dim3 ggrid(Dc / 128, (Bc * Tc) / 128);  // (8, 32)
  const dim3 agrid(Tc / 128, Bc * Hc);          // (8, 64)
  const int M = Bc * Tc;

  auto WL = [&](int wi, int l) { return Wb + wi * wN + (size_t)l * Dc * Dc; };

  for (int l = 0; l < Lc; l++) {
    const size_t lo = (size_t)l * Dc;
    // ---- self attention ----
    gemm_bf16<2><<<ggrid, 256, 0, stream>>>(xb, WL(0, l), Bsrc[0] + lo, nullptr, qb, M, Dc, Dc);
    gemm_bf16<2><<<ggrid, 256, 0, stream>>>(xb, WL(1, l), Bsrc[1] + lo, nullptr, kb, M, Dc, Dc);
    gemm_bf16<2><<<ggrid, 256, 0, stream>>>(xb, WL(2, l), Bsrc[2] + lo, nullptr, vb, M, Dc, Dc);
    flash_attn<true><<<agrid, 256, 0, stream>>>(qb, kb, vb, ab, Tc, Tc);
    gemm_bf16<1><<<ggrid, 256, 0, stream>>>(ab, WL(3, l), Bsrc[3] + lo, t0, nullptr, M, Dc, Dc);
    add_ln_kernel<<<M, 256, 0, stream>>>(x, t0, l1w + lo, l1b + lo, xb);
    // ---- cross attention ----
    gemm_bf16<2><<<ggrid, 256, 0, stream>>>(xb,   WL(4, l), Bsrc[4] + lo, nullptr, qb, M, Dc, Dc);
    gemm_bf16<2><<<ggrid, 256, 0, stream>>>(encb, WL(5, l), Bsrc[5] + lo, nullptr, kb, M, Dc, Dc);
    gemm_bf16<2><<<ggrid, 256, 0, stream>>>(encb, WL(6, l), Bsrc[6] + lo, nullptr, vb, M, Dc, Dc);
    flash_attn<false><<<agrid, 256, 0, stream>>>(qb, kb, vb, ab, Tc, Sc);
    gemm_bf16<1><<<ggrid, 256, 0, stream>>>(ab, WL(7, l), Bsrc[7] + lo, t0, nullptr, M, Dc, Dc);
    add_ln_kernel<<<M, 256, 0, stream>>>(x, t0, l2w + lo, l2b + lo, xb);
    // ---- FFN ----
    gemm_bf16<6><<<ggrid, 256, 0, stream>>>(xb, WL(8, l), Bsrc[8] + lo, nullptr, hb, M, Dc, Dc); // GELU, bf16
    gemm_bf16<1><<<ggrid, 256, 0, stream>>>(hb, WL(9, l), Bsrc[9] + lo, t0, nullptr, M, Dc, Dc);
    ffn_ln_kernel<<<M, 256, 0, stream>>>(x, t0, fnw + lo, fnb + lo, l3w + lo, l3b + lo, xb);
  }
}
